// MyParticleNetwork_88880053224118
// MI455X (gfx1250) — matrix operand order repaired and checked by reading, not compiled, not run on hardware
//
#include <hip/hip_runtime.h>
#include <stdint.h>

// ---------------------------------------------------------------------------
// Types for CDNA5 WMMA
// ---------------------------------------------------------------------------
typedef __attribute__((ext_vector_type(16))) __bf16 v16bf;
typedef __attribute__((ext_vector_type(8)))  float  v8f;

#define GCELLS 64   // KS^3 with KS=4

// ---------------------------------------------------------------------------
// bf16 helpers (round-to-nearest-even pack, raw unpack)
// ---------------------------------------------------------------------------
__device__ __host__ inline uint16_t f2bf(float f) {
    union { float f; uint32_t u; } v; v.f = f;
    uint32_t u = v.u;
    uint32_t r = u + 0x7fffu + ((u >> 16) & 1u);
    return (uint16_t)(r >> 16);
}
__device__ inline float bf2f(uint32_t h) {
    union { uint32_t u; float f; } v; v.u = h << 16; return v.f;
}
__device__ inline float sgnf(float v) { return (v > 0.f) ? 1.f : ((v < 0.f) ? -1.f : 0.f); }

// ---------------------------------------------------------------------------
// ball_to_cube (reference math, f32)
// ---------------------------------------------------------------------------
__device__ inline void ball2cube(float x, float y, float z,
                                 float& qx, float& qy, float& qz) {
    float sq  = x*x + y*y + z*z;
    float nrm = sqrtf(fmaxf(sq, 1e-12f));
    float xy  = x*x + y*y;
    bool  polar = 1.25f * z * z > xy;
    float sp = sqrtf(3.0f * nrm / (nrm + fabsf(z) + 1e-12f));
    float ss = nrm * rsqrtf(xy + 1e-12f);
    float s  = polar ? sp : ss;
    float cx = x * s, cy = y * s;
    float cz = polar ? sgnf(z) * nrm : 1.5f * z;
    if (sq < 1e-12f) { cx = 0.f; cy = 0.f; cz = 0.f; }
    float nxy = sqrtf(fmaxf(cx*cx + cy*cy, 1e-24f));
    bool  cxcase = fabsf(cx) >= fabsf(cy);
    float den = cxcase ? cx : cy;
    den = (fabsf(den) > 1e-12f) ? den : 1.0f;
    float r1 = cy / den, r2 = cx / den;
    const float c4pi = 1.27323954473516f;  // 4/pi
    float ax = sgnf(cx) * nxy;
    float ay = sgnf(cy) * nxy;
    qx = cxcase ? ax : ay * c4pi * atanf(r2);
    qy = cxcase ? ax * c4pi * atanf(r1) : ay;
    if (cx*cx + cy*cy < 1e-12f) { qx = 0.f; qy = 0.f; }
    qz = cz;
}

// ---------------------------------------------------------------------------
// Fused CConv (+ folded dense) kernel.
//   One workgroup = one 16-particle tile (WMMA M=16).
//   Phase 1: zero LDS A tile (16 x KA bf16, KA = 64*CINP (+CINP if FOLD)).
//   Phase 2: scatter neighbors into LDS with ds_pk_add_bf16 atomics.
//   Phase 3: (FOLD) copy query features into A tail rows (dense fold).
//   Phase 4: per-wave WMMA over K (dual accumulator chains for XDL ILP),
//            epilogue: +bias (+resid), *scale.
// ---------------------------------------------------------------------------
static constexpr size_t cconvShmemBytes(int CINP, bool FOLD) {
    size_t KA = (size_t)GCELLS * CINP + (FOLD ? CINP : 0);
    return 16 * (KA + 8) * 2;  // bf16, +8 halves row pad vs bank conflicts
}

union ABFrag { uint4 q[2]; v16bf v; };

template <int CINP, int COUT, int COUTP, bool FOLD, bool RES>
__global__ __launch_bounds__(128) void cconv_wmma(
    const float*    __restrict__ posq, int Nq,
    const float*    __restrict__ poss,
    const uint32_t* __restrict__ featPairs,   // bf16x2, CINP/2 per source row
    const int*      __restrict__ nbr,
    const uint8_t*  __restrict__ mask, int K,
    const uint16_t* __restrict__ Wp,          // packed B fragments (bf16)
    const float*    __restrict__ bias0,
    const float*    __restrict__ bias1,       // nullable (dense bias)
    const float*    __restrict__ resid,       // nullable
    float*          __restrict__ out, int outStride, int outOffset, float scale)
{
    constexpr int KA = GCELLS * CINP + (FOLD ? CINP : 0);
    constexpr int RS = KA + 8;               // row stride in halves
    constexpr int KT = KA / 32;              // number of 16x16x32 k-tiles
    extern __shared__ uint16_t smem[];

    const int tid   = threadIdx.x;
    const int tile0 = blockIdx.x * 16;

    // ---- Phase 1: zero A tile ----
    {
        uint4 z; z.x = z.y = z.z = z.w = 0u;
        uint4* p4 = (uint4*)smem;
        const int n4 = 16 * RS / 8;
        for (int i = tid; i < n4; i += blockDim.x) p4[i] = z;
    }
    __syncthreads();

    // ---- Phase 2: scatter neighbors with ds_pk_add_bf16 ----
    const float invR = 1.0f / (0.5f * (1.5f * 6.0f * 0.025f));   // 1/RADIUS
    for (int t = tid; t < 16 * K; t += blockDim.x) {
        int row = t / K, kk = t - row * K;
        int q = tile0 + row;
        if (q >= Nq) continue;
        int   idx = nbr[q * K + kk];
        float m   = (float)mask[q * K + kk];

        float qx = posq[q * 3 + 0], qy = posq[q * 3 + 1], qz = posq[q * 3 + 2];
        float px = (poss[idx * 3 + 0] - qx) * invR;
        float py = (poss[idx * 3 + 1] - qy) * invR;
        float pz = (poss[idx * 3 + 2] - qz) * invR;
        float r2 = px * px + py * py + pz * pz;

        float ux, uy, uz;
        ball2cube(px, py, pz, ux, uy, uz);
        float gx = (ux + 1.0f) * 1.5f;
        float gy = (uy + 1.0f) * 1.5f;
        float gz = (uz + 1.0f) * 1.5f;
        float fx = fminf(fmaxf(floorf(gx), 0.f), 2.f);
        float fy = fminf(fmaxf(floorf(gy), 0.f), 2.f);
        float fz = fminf(fmaxf(floorf(gz), 0.f), 2.f);
        float tx = gx - fx, ty = gy - fy, tz = gz - fz;
        int ix = (int)fx, iy = (int)fy, iz = (int)fz;

        float win = 1.0f - r2;
        win = win * win * win;
        win = fminf(fmaxf(win, 0.f), 1.f) * m;

        float wxl[2] = { 1.f - tx, tx };
        float wyl[2] = { 1.f - ty, ty };
        float wzl[2] = { 1.f - tz, tz };
        float cw[8]; int cc[8];
        #pragma unroll
        for (int dz = 0; dz < 2; ++dz)
            #pragma unroll
            for (int dy = 0; dy < 2; ++dy)
                #pragma unroll
                for (int dx = 0; dx < 2; ++dx) {
                    int j = dz * 4 + dy * 2 + dx;
                    cw[j] = win * wzl[dz] * wyl[dy] * wxl[dx];
                    cc[j] = ((iz + dz) * 4 + (iy + dy)) * 4 + (ix + dx);
                }

        uint32_t abase = (uint32_t)(uintptr_t)smem + (uint32_t)(row * RS) * 2u;
        const uint32_t* fp = featPairs + (size_t)idx * (CINP / 2);
        for (int c2 = 0; c2 < CINP / 2; ++c2) {
            uint32_t pr = fp[c2];
            float f0 = bf2f(pr & 0xffffu);
            float f1 = bf2f(pr >> 16);
            #pragma unroll
            for (int j = 0; j < 8; ++j) {
                float w = cw[j];
                uint32_t pk = ((uint32_t)f2bf(w * f1) << 16) | (uint32_t)f2bf(w * f0);
                uint32_t addr = abase + (uint32_t)(cc[j] * CINP + 2 * c2) * 2u;
                asm volatile("ds_pk_add_bf16 %0, %1" :: "v"(addr), "v"(pk) : "memory");
            }
        }
    }

    // ---- Phase 3: fold dense: copy query features into A tail ----
    if (FOLD) {
        for (int t = tid; t < 16 * (CINP / 2); t += blockDim.x) {
            int row = t / (CINP / 2), c2 = t - row * (CINP / 2);
            int q = tile0 + row;
            uint32_t v = (q < Nq) ? featPairs[(size_t)q * (CINP / 2) + c2] : 0u;
            *(uint32_t*)(smem + row * RS + GCELLS * CINP + 2 * c2) = v;
        }
    }
    asm volatile("s_wait_dscnt 0" ::: "memory");
    __syncthreads();

    // ---- Phase 4: WMMA GEMM  C[16 x 16] += A[16 x KA] * Wp[KA x 16] ----
    const int wv = tid >> 5, lane = tid & 31;
    constexpr int NT = COUTP / 16;
    if (wv < NT) {
        const int nt = wv;
        v8f acc0 = {}, acc1 = {};
        // B fragment pointer: per (nt,kt,lane) lane owns 16 contiguous halves.
        const uint4* wp = (const uint4*)Wp + ((size_t)nt * KT) * 64 + lane * 2;
        const int row  = lane & 15;
        const int koff = (lane < 16) ? 0 : 8;   // ISA A-matrix 16-bit layout
        const uint16_t* arow = smem + row * RS;
        // Dual independent accumulator chains: hides XDL latency at the
        // 1-wave/SIMD occupancy this LDS footprint implies.
        int kt = 0;
        for (; kt + 1 < KT; kt += 2) {
            ABFrag a0, b0, a1, b1;
            b0.q[0] = wp[0];
            b0.q[1] = wp[1];
            b1.q[0] = wp[64];
            b1.q[1] = wp[65];
            int k0 = kt * 32 + koff;
            a0.q[0] = *(const uint4*)(arow + k0);
            a0.q[1] = *(const uint4*)(arow + k0 + 16);
            a1.q[0] = *(const uint4*)(arow + k0 + 32);
            a1.q[1] = *(const uint4*)(arow + k0 + 48);
            wp += 128;
            __builtin_prefetch(wp, 0, 3);       // global_prefetch_b8 next B tiles
            acc0 = __builtin_amdgcn_wmma_f32_16x16x32_bf16(
                false, a0.v, false, b0.v, (short)0, acc0, false, false);
            acc1 = __builtin_amdgcn_wmma_f32_16x16x32_bf16(
                false, a1.v, false, b1.v, (short)0, acc1, false, false);
        }
        if (KT & 1) {                            // tail k-tile (conv1: KT=195)
            ABFrag a0, b0;
            b0.q[0] = wp[0];
            b0.q[1] = wp[1];
            int k0 = kt * 32 + koff;
            a0.q[0] = *(const uint4*)(arow + k0);
            a0.q[1] = *(const uint4*)(arow + k0 + 16);
            acc0 = __builtin_amdgcn_wmma_f32_16x16x32_bf16(
                false, a0.v, false, b0.v, (short)0, acc0, false, false);
        }
        v8f acc = acc0 + acc1;
        // Epilogue: C layout VGPR r -> row = r + (lane<16?0:8), col = lane&15
        const int col   = lane & 15;
        const int rbase = (lane < 16) ? 0 : 8;
        const int oc    = nt * 16 + col;
        if (oc < COUT) {
            float bs = bias0[oc] + (bias1 ? bias1[oc] : 0.0f);
            #pragma unroll
            for (int r = 0; r < 8; ++r) {
                int q = tile0 + rbase + r;
                if (q < Nq) {
                    float v = acc[r] + bs;
                    if (RES) v += resid[(size_t)q * outStride + outOffset + oc];
                    out[(size_t)q * outStride + outOffset + oc] = v * scale;
                }
            }
        }
    }
}

// ---------------------------------------------------------------------------
// Weight packer: (G,Cin,Cout) f32 [+ dense (Cin,Cout)] -> B-fragment bf16.
// Fragment layout (per kt,nt): lane 0-15 hold K=kt*32+{0..7,16..23} for col=lane,
// lanes 16-31 hold K=kt*32+{8..15,24..31} for col=lane-16; 16 halves contiguous/lane.
// ---------------------------------------------------------------------------
__global__ void pack_w_kernel(const float* __restrict__ W,
                              const float* __restrict__ D,
                              uint16_t* __restrict__ Wp,
                              int Cin, int Cout, int CINP, int KA, int nTiles)
{
    int t = blockIdx.x * blockDim.x + threadIdx.x;
    int total = nTiles * (KA / 32) * 32;
    if (t >= total) return;
    int lane  = t & 31;
    int kt    = (t >> 5) % (KA / 32);
    int nt    = (t >> 5) / (KA / 32);
    int n     = nt * 16 + (lane & 15);
    int kbase = kt * 32 + ((lane >> 4) << 3);
    union { uint16_t h[16]; uint4 q[2]; } u;
    #pragma unroll
    for (int j = 0; j < 16; ++j) {
        int k = kbase + (j < 8 ? j : j + 8);
        float v = 0.f;
        int g = k / CINP, c = k - g * CINP;
        if (n < Cout && c < Cin) {
            if (g < GCELLS)        v = W[((size_t)(g * Cin + c)) * Cout + n];
            else if (D != nullptr) v = D[(size_t)c * Cout + n];
        }
        u.h[j] = f2bf(v);
    }
    uint4* dq = (uint4*)(Wp + (size_t)t * 16);
    dq[0] = u.q[0];
    dq[1] = u.q[1];
}

// ---------------------------------------------------------------------------
// Small prep kernels
// ---------------------------------------------------------------------------
__global__ void prep_fluid_feats(const float* __restrict__ vel,
                                 uint32_t* __restrict__ flb, int N) {
    int i = blockIdx.x * blockDim.x + threadIdx.x;
    if (i >= N) return;
    flb[i * 2 + 0] = ((uint32_t)f2bf(vel[i * 3 + 0]) << 16) | (uint32_t)f2bf(1.0f);
    flb[i * 2 + 1] = ((uint32_t)f2bf(vel[i * 3 + 2]) << 16) | (uint32_t)f2bf(vel[i * 3 + 1]);
}
__global__ void prep_box_feats(const float* __restrict__ bf,
                               uint32_t* __restrict__ bb, int M) {
    int i = blockIdx.x * blockDim.x + threadIdx.x;
    if (i >= M) return;
    bb[i * 2 + 0] = ((uint32_t)f2bf(bf[i * 3 + 1]) << 16) | (uint32_t)f2bf(bf[i * 3 + 0]);
    bb[i * 2 + 1] = (uint32_t)f2bf(bf[i * 3 + 2]);   // channel 3 zero pad
}
__global__ void dense0_kernel(const float* __restrict__ vel,
                              const float* __restrict__ W,  // (4,32)
                              const float* __restrict__ b,
                              float* __restrict__ ans, int N) {
    int t = blockIdx.x * blockDim.x + threadIdx.x;
    if (t >= N * 32) return;
    int n = t >> 5, o = t & 31;
    float v = W[o] + vel[n * 3 + 0] * W[32 + o] + vel[n * 3 + 1] * W[64 + o]
                   + vel[n * 3 + 2] * W[96 + o] + b[o];
    ans[(size_t)n * 96 + 64 + o] = v;
}
__global__ void relu_bf16_kernel(const float* __restrict__ x,
                                 uint32_t* __restrict__ xb, int nPairs) {
    int i = blockIdx.x * blockDim.x + threadIdx.x;
    if (i >= nPairs) return;
    float v0 = fmaxf(x[2 * i + 0], 0.f);
    float v1 = fmaxf(x[2 * i + 1], 0.f);
    xb[i] = ((uint32_t)f2bf(v1) << 16) | (uint32_t)f2bf(v0);
}

// ---------------------------------------------------------------------------
// Host launcher
// ---------------------------------------------------------------------------
extern "C" void kernel_launch(void* const* d_in, const int* in_sizes, int n_in,
                              void* d_out, int out_size, void* d_ws, size_t ws_size,
                              hipStream_t stream) {
    (void)n_in; (void)out_size; (void)ws_size;
    const float* pos   = (const float*)d_in[0];
    const float* vel   = (const float*)d_in[1];
    const float* box   = (const float*)d_in[2];
    const float* boxf  = (const float*)d_in[3];
    const float* c0fw  = (const float*)d_in[4];
    const float* c0fb  = (const float*)d_in[5];
    const float* c0ow  = (const float*)d_in[6];
    const float* c0ob  = (const float*)d_in[7];
    const float* d0w   = (const float*)d_in[8];
    const float* d0b   = (const float*)d_in[9];
    const float* c1w   = (const float*)d_in[10];
    const float* c1b   = (const float*)d_in[11];
    const float* d1w   = (const float*)d_in[12];
    const float* d1b   = (const float*)d_in[13];
    const float* c2w   = (const float*)d_in[14];
    const float* c2b   = (const float*)d_in[15];
    const float* d2w   = (const float*)d_in[16];
    const float* d2b   = (const float*)d_in[17];
    const float* c3w   = (const float*)d_in[18];
    const float* c3b   = (const float*)d_in[19];
    const float* d3w   = (const float*)d_in[20];
    const float* d3b   = (const float*)d_in[21];
    const int*     fidx  = (const int*)d_in[22];
    const uint8_t* fmask = (const uint8_t*)d_in[23];
    const int*     bidx  = (const int*)d_in[24];
    const uint8_t* bmask = (const uint8_t*)d_in[25];

    const int N  = in_sizes[0] / 3;
    const int M  = in_sizes[2] / 3;
    const int Kf = in_sizes[22] / N;
    const int Kb = in_sizes[24] / N;

    // workspace carve
    char* base = (char*)d_ws;
    size_t off = 0;
    auto take = [&](size_t bytes) -> char* {
        char* r = base + off;
        off += (bytes + 255) & ~(size_t)255;
        return r;
    };
    float*    ans96 = (float*)take((size_t)N * 96 * 4);
    float*    ansB  = (float*)take((size_t)N * 64 * 4);
    uint32_t* xb    = (uint32_t*)take((size_t)N * 96 * 2);
    uint32_t* flb   = (uint32_t*)take((size_t)N * 4 * 2);
    uint32_t* bfb   = (uint32_t*)take((size_t)M * 4 * 2);
    uint16_t* Wp0f  = (uint16_t*)take((size_t)256 * 32 * 2);
    uint16_t* Wp0o  = (uint16_t*)take((size_t)256 * 32 * 2);
    uint16_t* Wp1   = (uint16_t*)take((size_t)6240 * 64 * 2);
    uint16_t* Wp2   = (uint16_t*)take((size_t)4160 * 64 * 2);
    uint16_t* Wp3   = (uint16_t*)take((size_t)4160 * 16 * 2);

    const int tiles = (N + 15) / 16;

    // feature prep
    prep_fluid_feats<<<(N + 255) / 256, 256, 0, stream>>>(vel, flb, N);
    prep_box_feats  <<<(M + 255) / 256, 256, 0, stream>>>(boxf, bfb, M);

    // weight packing (conv [+ folded dense] -> B fragments)
    auto pack = [&](const float* W, const float* D, uint16_t* Wp,
                    int Cin, int Cout, int CINP, int KA, int nT) {
        int tot = nT * (KA / 32) * 32;
        pack_w_kernel<<<(tot + 255) / 256, 256, 0, stream>>>(W, D, Wp, Cin, Cout, CINP, KA, nT);
    };
    pack(c0ow, nullptr, Wp0o, 3, 32, 4, 256, 2);
    pack(c0fw, nullptr, Wp0f, 4, 32, 4, 256, 2);
    pack(c1w,  d1w,     Wp1, 96, 64, 96, 6240, 4);
    pack(c2w,  d2w,     Wp2, 64, 64, 64, 4160, 4);
    pack(c3w,  d3w,     Wp3, 64,  3, 64, 4160, 1);

    // layer 0: ans = [conv0_obstacle | conv0_fluid | dense0]
    cconv_wmma<4, 32, 32, false, false>
        <<<tiles, 128, cconvShmemBytes(4, false), stream>>>(
            pos, N, box, bfb, bidx, bmask, Kb, Wp0o, c0ob, nullptr, nullptr,
            ans96, 96, 0, 1.0f);
    cconv_wmma<4, 32, 32, false, false>
        <<<tiles, 128, cconvShmemBytes(4, false), stream>>>(
            pos, N, pos, flb, fidx, fmask, Kf, Wp0f, c0fb, nullptr, nullptr,
            ans96, 96, 32, 1.0f);
    dense0_kernel<<<((size_t)N * 32 + 255) / 256, 256, 0, stream>>>(vel, d0w, d0b, ans96, N);

    // layer 1: ansB = conv1(relu(ans)) + dense1(relu(ans))   (dense folded)
    relu_bf16_kernel<<<((size_t)N * 48 + 255) / 256, 256, 0, stream>>>(ans96, xb, N * 48);
    cconv_wmma<96, 64, 64, true, false>
        <<<tiles, 128, cconvShmemBytes(96, true), stream>>>(
            pos, N, pos, xb, fidx, fmask, Kf, Wp1, c1b, d1b, nullptr,
            ansB, 64, 0, 1.0f);

    // layer 2: ansB = conv2(relu(ansB)) + dense2(...) + ansB  (residual)
    relu_bf16_kernel<<<((size_t)N * 32 + 255) / 256, 256, 0, stream>>>(ansB, xb, N * 32);
    cconv_wmma<64, 64, 64, true, true>
        <<<tiles, 128, cconvShmemBytes(64, true), stream>>>(
            pos, N, pos, xb, fidx, fmask, Kf, Wp2, c2b, d2b, ansB,
            ansB, 64, 0, 1.0f);

    // layer 3: out = (conv3(relu(ansB)) + dense3(...)) / 128
    relu_bf16_kernel<<<((size_t)N * 32 + 255) / 256, 256, 0, stream>>>(ansB, xb, N * 32);
    cconv_wmma<64, 3, 16, true, false>
        <<<tiles, 128, cconvShmemBytes(64, true), stream>>>(
            pos, N, pos, xb, fidx, fmask, Kf, Wp3, c3b, d3b, nullptr,
            (float*)d_out, 3, 0, 1.0f / 128.0f);
}
